// Knowledge_Manager_43954695307866
// MI455X (gfx1250) — matrix operand order repaired and checked by reading, not compiled, not run on hardware
//
#include <hip/hip_runtime.h>
#include <hip/hip_bf16.h>
#include <math.h>
#include <stdint.h>

// Problem constants (match reference)
#define B_   32
#define N_   32
#define XS_  256
#define YS_  256
#define KS_  128
#define D_   1024
#define V_   32000
#define TAU_INV 1.0e4f

typedef __attribute__((ext_vector_type(16))) __bf16        v16bf;
typedef __attribute__((ext_vector_type(8)))  __bf16        v8bf;
typedef __attribute__((ext_vector_type(8)))  float         v8f;
typedef __attribute__((ext_vector_type(4)))  unsigned int  u32x4;
typedef __attribute__((ext_vector_type(8)))  int           i32x8;
typedef __attribute__((ext_vector_type(4)))  int           i32x4;

__device__ __forceinline__ float wave_max_f(float v) {
#pragma unroll
  for (int o = 16; o > 0; o >>= 1) v = fmaxf(v, __shfl_xor(v, o, 32));
  return v;
}
__device__ __forceinline__ float wave_sum_f(float v) {
#pragma unroll
  for (int o = 16; o > 0; o >>= 1) v += __shfl_xor(v, o, 32);
  return v;
}

// ---------------------------------------------------------------------------
// TDM: DMA a 2-D tile (rows x elems_per_row) from global memory into LDS.
// Descriptor bitfields packed per CDNA5 ISA ch.8 (groups 0..3).
// data_size_code: 0=1B, 1=2B, 2=4B, 3=8B.
// ---------------------------------------------------------------------------
__device__ __forceinline__ void tdm_load_2d_to_lds(const void* gaddr, unsigned lds_off,
                                                   unsigned elems_per_row, unsigned rows,
                                                   unsigned row_stride_elems,
                                                   unsigned data_size_code) {
  const unsigned long long ga = (unsigned long long)(uintptr_t)gaddr;
  u32x4 g0;
  g0.x = 1u;                                   // count=1 (valid user descriptor)
  g0.y = lds_off;                              // lds_addr (bytes)
  g0.z = (unsigned)ga;                         // global_addr[31:0]
  g0.w = (unsigned)((ga >> 32) & 0x1FFFFFFu)   // global_addr[56:32]
         | (2u << 30);                         // type=2 ("image")
  i32x8 g1;
  g1[0] = (int)(data_size_code << 16);                         // data_size; no flags/pad
  g1[1] = (int)((elems_per_row & 0xFFFFu) << 16);              // tensor_dim0[15:0]
  g1[2] = (int)(((elems_per_row >> 16) & 0xFFFFu)              // tensor_dim0[31:16]
                | ((rows & 0xFFFFu) << 16));                   // tensor_dim1[15:0]
  g1[3] = (int)(((rows >> 16) & 0xFFFFu)                       // tensor_dim1[31:16]
                | ((elems_per_row & 0xFFFFu) << 16));          // tile_dim0
  g1[4] = (int)(rows & 0xFFFFu);                               // tile_dim1 (tile_dim2=0)
  g1[5] = (int)row_stride_elems;                               // tensor_dim0_stride[31:0]
  g1[6] = 0;                                                   // stride0[47:32], stride1 lo
  g1[7] = 0;
  i32x4 gz = {0, 0, 0, 0};                                     // 2-D tile: groups 2/3 unused
#if __clang_major__ >= 23
  i32x8 gz8 = {0, 0, 0, 0, 0, 0, 0, 0};
  __builtin_amdgcn_tensor_load_to_lds(g0, g1, gz, gz, gz8, 0);
#else
  __builtin_amdgcn_tensor_load_to_lds(g0, g1, gz, gz, 0);
#endif
}

// ---------------------------------------------------------------------------
// Vectorized WMMA fragment loaders (layouts per CDNA5 ISA 7.12.2, wave32):
//  A 16x32 (MxK) bf16: lane<16: e0..7->K0..7, e8..15->K16..23 (M=lane&15);
//                      lane>=16: +8 on K.
//  B 32x16 (KxN) bf16: N=lane&15; same K layout as A.
//  C/D 16x16 f32: VGPR r -> M = r + 8*(lane>=16), N = lane&15.
// ---------------------------------------------------------------------------
__device__ __forceinline__ v16bf load_a_f32(const float* __restrict__ p) {
  // p points at K-run start (k0 + kA): e0..7 = p[0..7], e8..15 = p[16..23]
  const float4 r0 = *(const float4*)(p);
  const float4 r1 = *(const float4*)(p + 4);
  const float4 r2 = *(const float4*)(p + 16);
  const float4 r3 = *(const float4*)(p + 20);
  v16bf a;
  a[0]=(__bf16)r0.x;  a[1]=(__bf16)r0.y;  a[2]=(__bf16)r0.z;  a[3]=(__bf16)r0.w;
  a[4]=(__bf16)r1.x;  a[5]=(__bf16)r1.y;  a[6]=(__bf16)r1.z;  a[7]=(__bf16)r1.w;
  a[8]=(__bf16)r2.x;  a[9]=(__bf16)r2.y;  a[10]=(__bf16)r2.z; a[11]=(__bf16)r2.w;
  a[12]=(__bf16)r3.x; a[13]=(__bf16)r3.y; a[14]=(__bf16)r3.z; a[15]=(__bf16)r3.w;
  return a;
}
__device__ __forceinline__ v16bf load_a_ldsbf(const __bf16* p) {
  const v8bf lo = *(const v8bf*)(p);        // K +0..7   (ds_load_b128)
  const v8bf hi = *(const v8bf*)(p + 16);   // K +16..23 (ds_load_b128)
  v16bf a;
#pragma unroll
  for (int e = 0; e < 8; ++e) { a[e] = lo[e]; a[e + 8] = hi[e]; }
  return a;
}
__device__ __forceinline__ v16bf load_b_f32(const float* __restrict__ p) {
  // p points at K-run start (k0 + kB): e0..7 = p[0..7], e8..15 = p[16..23]
  const float4 q0 = *(const float4*)(p);
  const float4 q1 = *(const float4*)(p + 4);
  const float4 q2 = *(const float4*)(p + 16);
  const float4 q3 = *(const float4*)(p + 20);
  v16bf b;
  b[0]=(__bf16)q0.x;  b[1]=(__bf16)q0.y;  b[2]=(__bf16)q0.z;  b[3]=(__bf16)q0.w;
  b[4]=(__bf16)q1.x;  b[5]=(__bf16)q1.y;  b[6]=(__bf16)q1.z;  b[7]=(__bf16)q1.w;
  b[8]=(__bf16)q2.x;  b[9]=(__bf16)q2.y;  b[10]=(__bf16)q2.z; b[11]=(__bf16)q2.w;
  b[12]=(__bf16)q3.x; b[13]=(__bf16)q3.y; b[14]=(__bf16)q3.z; b[15]=(__bf16)q3.w;
  return b;
}

// ---------------------------------------------------------------------------
// K1: masked mean over sequence for utterance (blockIdx.y==0) / response (==1)
// ---------------------------------------------------------------------------
__global__ void k_mean_xy(const float* __restrict__ uh, const float* __restrict__ rh,
                          const unsigned char* __restrict__ um,
                          const unsigned char* __restrict__ rm,
                          float* __restrict__ meanX, float* __restrict__ meanY) {
  const int b   = blockIdx.x;
  const bool isY = (blockIdx.y != 0);
  const float* src = isY ? rh : uh;
  const unsigned char* msk = isY ? rm : um;
  float* dst = isY ? meanY : meanX;

  __shared__ unsigned char keep[XS_];
  keep[threadIdx.x] = msk[b * XS_ + threadIdx.x] ? 0 : 1;  // blockDim==XS_==YS_
  __syncthreads();

  const int d = threadIdx.x * 4;
  float ax = 0.f, ay = 0.f, az = 0.f, aw = 0.f;
  for (int x = 0; x < XS_; ++x) {
    if (keep[x]) {
      const float4 v = *(const float4*)(src + ((size_t)x * B_ + b) * D_ + d);
      ax += v.x; ay += v.y; az += v.z; aw += v.w;
    }
  }
  const float inv = 1.0f / (float)XS_;
  float4 o; o.x = ax * inv; o.y = ay * inv; o.z = az * inv; o.w = aw * inv;
  *(float4*)(dst + (size_t)b * D_ + d) = o;
}

// ---------------------------------------------------------------------------
// K2: mean_K[b,n,d] = (1/KS) * sum_s (~mask) * kh[n,s,b,d].  512 MB stream.
// ---------------------------------------------------------------------------
__global__ void k_mean_k(const float* __restrict__ kh,
                         const unsigned char* __restrict__ km,
                         float* __restrict__ meanK) {
  const int n = blockIdx.x, b = blockIdx.y;
  __shared__ unsigned char keep[KS_];
  if (threadIdx.x < KS_)
    keep[threadIdx.x] = km[((size_t)b * N_ + n) * KS_ + threadIdx.x] ? 0 : 1;
  __syncthreads();

  const int d = threadIdx.x * 4;
  float ax = 0.f, ay = 0.f, az = 0.f, aw = 0.f;
  for (int s = 0; s < KS_; ++s) {
    const float* p = kh + (((size_t)n * KS_ + s) * B_ + b) * D_ + d;
    if (s + 1 < KS_) __builtin_prefetch(p + (size_t)B_ * D_, 0, 1);  // global_prefetch_b8
    if (keep[s]) {
      const float4 v = *(const float4*)p;
      ax += v.x; ay += v.y; az += v.z; aw += v.w;
    }
  }
  const float inv = 1.0f / (float)KS_;
  float4 o; o.x = ax * inv; o.y = ay * inv; o.z = az * inv; o.w = aw * inv;
  *(float4*)(meanK + ((size_t)b * N_ + n) * D_ + d) = o;
}

// ---------------------------------------------------------------------------
// K3: XYproj = [mean_X|mean_Y] @ W1^T + b1.  M=32, K=2048, N=1024 (WMMA bf16).
// grid 8 blocks x 8 waves; wave -> one 16-wide N tile, both M tiles.
// ---------------------------------------------------------------------------
__global__ void k_w1_gemm(const float* __restrict__ meanX, const float* __restrict__ meanY,
                          const float* __restrict__ W1, const float* __restrict__ b1,
                          float* __restrict__ XYproj) {
  const int lane = threadIdx.x & 31;
  const int wave = threadIdx.x >> 5;
  const int n0   = (blockIdx.x * 8 + wave) * 16;       // 64 tiles total
  const int mrow = lane & 15;
  const int kA   = (lane >> 4) << 3;                   // 0 or 8
  const int kB   = (lane >> 4) << 3;                   // 0 or 8
  const int ncol = n0 + (lane & 15);

  v8f acc0 = {0.f,0.f,0.f,0.f,0.f,0.f,0.f,0.f};
  v8f acc1 = {0.f,0.f,0.f,0.f,0.f,0.f,0.f,0.f};
  const float* brow = W1 + (size_t)ncol * (2 * D_);

  for (int k0 = 0; k0 < 2 * D_; k0 += 32) {
    // whole 32-wide k-step lies entirely in meanX (k0<1024) or meanY half
    const float* half = (k0 < D_) ? meanX : (meanY - D_);
    const v16bf a0  = load_a_f32(half + (size_t)mrow * D_ + k0 + kA);
    const v16bf a1  = load_a_f32(half + (size_t)(mrow + 16) * D_ + k0 + kA);
    const v16bf bfr = load_b_f32(brow + k0 + kB);
    acc0 = __builtin_amdgcn_wmma_f32_16x16x32_bf16(false, a0, false, bfr, (short)0, acc0, false, false);
    acc1 = __builtin_amdgcn_wmma_f32_16x16x32_bf16(false, a1, false, bfr, (short)0, acc1, false, false);
  }

  const float bias  = b1[ncol];
  const int   mbase = (lane >> 4) * 8;
#pragma unroll
  for (int r = 0; r < 8; ++r) {
    const int m = mbase + r;
    XYproj[(size_t)m * D_ + ncol]        = acc0[r] + bias;
    XYproj[(size_t)(m + 16) * D_ + ncol] = acc1[r] + bias;
  }
}

// ---------------------------------------------------------------------------
// K4: per-batch logits + three softmaxes + vocab_selected pre-GEMM vector
//     (written as bf16, ready for TDM staging + WMMA A fragments).
// ---------------------------------------------------------------------------
__global__ void k_softmaxes(const float* __restrict__ meanX, const float* __restrict__ XYproj,
                            const float* __restrict__ meanK, const float* __restrict__ gumbel,
                            float* __restrict__ o_prior, float* __restrict__ o_post,
                            float* __restrict__ o_kidx, float* __restrict__ ws_kidx,
                            __bf16* __restrict__ Svbf) {
  const int b    = blockIdx.x;
  const int lane = threadIdx.x & 31;
  const int wave = threadIdx.x >> 5;
  __shared__ float sp[N_], pp[N_], ki[N_];

#pragma unroll
  for (int j = 0; j < 4; ++j) {
    const int n = wave * 4 + j;
    const float* mk = meanK + ((size_t)b * N_ + n) * D_;
    const float* mx = meanX + (size_t)b * D_;
    const float* xy = XYproj + (size_t)b * D_;
    float a = 0.f, c = 0.f;
    for (int d = lane; d < D_; d += 32) {
      const float kv = mk[d];
      a += mx[d] * kv;
      c += xy[d] * kv;
    }
    a = wave_sum_f(a);
    c = wave_sum_f(c);
    if (lane == 0) { sp[n] = a; pp[n] = c; }
  }
  __syncthreads();

  if (threadIdx.x < 32) {              // whole wave 0: EXEC fully on
    const int n = threadIdx.x;
    // prior = log_softmax(sp)
    const float sl = sp[n];
    const float m1 = wave_max_f(sl);
    const float s1 = wave_sum_f(expf(sl - m1));
    o_prior[b * N_ + n] = sl - m1 - logf(s1);
    // posterior = softmax(pp)
    const float pl = pp[n];
    const float m2 = wave_max_f(pl);
    const float e2 = expf(pl - m2);
    const float s2 = wave_sum_f(e2);
    o_post[b * N_ + n] = e2 / s2;
    // K_index = softmax((pp + gumbel)/tau) -> numerically one-hot-ish
    const float gl = (pl + gumbel[b * N_ + n]) * TAU_INV;
    const float m3 = wave_max_f(gl);
    const float e3 = expf(gl - m3);
    const float s3 = wave_sum_f(e3);
    const float kv = e3 / s3;
    o_kidx[b * N_ + n] = kv;
    ws_kidx[b * N_ + n] = kv;
    ki[n] = kv;
  }
  __syncthreads();

  // Svbf[b,d] = bf16( sum_n ki[n] * mean_K[b,n,d] )   (skip exact zeros)
  const int d = threadIdx.x * 4;
  float ax = 0.f, ay = 0.f, az = 0.f, aw = 0.f;
  for (int n = 0; n < N_; ++n) {
    const float c = ki[n];
    if (c != 0.0f) {
      const float4 v = *(const float4*)(meanK + ((size_t)b * N_ + n) * D_ + d);
      ax += c * v.x; ay += c * v.y; az += c * v.z; aw += c * v.w;
    }
  }
  __bf16* o = Svbf + (size_t)b * D_ + d;
  o[0] = (__bf16)ax; o[1] = (__bf16)ay; o[2] = (__bf16)az; o[3] = (__bf16)aw;
}

// ---------------------------------------------------------------------------
// K5: selected_K[b,s,d] = sum_n (~mask[b,n,s]) * ki[b,n] * kh[n,s,b,d].
// Zero coefficients skip the 4 KB read (bit-exact: 0*finite contributes 0).
// ---------------------------------------------------------------------------
__global__ void k_selected(const float* __restrict__ kh,
                           const unsigned char* __restrict__ km,
                           const float* __restrict__ kidx,
                           float* __restrict__ o_sel) {
  const int s = blockIdx.x, b = blockIdx.y;
  __shared__ float coef[N_];
  if (threadIdx.x < N_) {
    const int n = threadIdx.x;
    const unsigned char m = km[((size_t)b * N_ + n) * KS_ + s];
    coef[n] = m ? 0.0f : kidx[b * N_ + n];
  }
  __syncthreads();

  const int d = threadIdx.x * 4;
  float ax = 0.f, ay = 0.f, az = 0.f, aw = 0.f;
  for (int n = 0; n < N_; ++n) {
    const float c = coef[n];
    if (c != 0.0f) {
      const float4 v = *(const float4*)(kh + (((size_t)n * KS_ + s) * B_ + b) * D_ + d);
      ax += c * v.x; ay += c * v.y; az += c * v.z; aw += c * v.w;
    }
  }
  float4 o; o.x = ax; o.y = ay; o.z = az; o.w = aw;
  *(float4*)(o_sel + ((size_t)b * KS_ + s) * D_ + d) = o;
}

// ---------------------------------------------------------------------------
// K6: Vlog = Svoc @ W2^T + b2.  M=32, K=1024, N=32000 (WMMA bf16, f32 acc).
// Per block: one TDM DMA stages the whole bf16 A matrix (32x1024 = 64 KB)
// into LDS; 8 waves each own a 16-wide N tile (both M tiles), streaming W2
// (128 MB) exactly once with b128 loads. A frags come from LDS (b128).
// ---------------------------------------------------------------------------
__global__ void k_w2_gemm(const __bf16* __restrict__ Svbf, const float* __restrict__ W2,
                          const float* __restrict__ b2, float* __restrict__ Vlog) {
  __shared__ __bf16 Alds[B_ * D_];           // 64 KB staged via tensor_load_to_lds
  const int lane = threadIdx.x & 31;
  const int wave = threadIdx.x >> 5;

  if (wave == 0) {
    tdm_load_2d_to_lds(Svbf, (unsigned)(uintptr_t)(&Alds[0]),
                       /*elems_per_row=*/D_, /*rows=*/B_,
                       /*row_stride=*/D_, /*data_size=2B*/1u);
    __builtin_amdgcn_s_wait_tensorcnt(0);
  }
  __syncthreads();

  const int n0   = (blockIdx.x * 8 + wave) * 16;       // 2000 tiles total
  const int mrow = lane & 15;
  const int kA   = (lane >> 4) << 3;
  const int kB   = (lane >> 4) << 3;
  const int ncol = n0 + (lane & 15);

  v8f acc0 = {0.f,0.f,0.f,0.f,0.f,0.f,0.f,0.f};
  v8f acc1 = {0.f,0.f,0.f,0.f,0.f,0.f,0.f,0.f};
  const __bf16* arow0 = Alds + (size_t)mrow * D_;
  const __bf16* arow1 = Alds + (size_t)(mrow + 16) * D_;
  const float*  brow  = W2 + (size_t)ncol * D_;

  for (int k0 = 0; k0 < D_; k0 += 32) {
    const v16bf a0  = load_a_ldsbf(arow0 + k0 + kA);
    const v16bf a1  = load_a_ldsbf(arow1 + k0 + kA);
    const v16bf bfr = load_b_f32(brow + k0 + kB);
    acc0 = __builtin_amdgcn_wmma_f32_16x16x32_bf16(false, a0, false, bfr, (short)0, acc0, false, false);
    acc1 = __builtin_amdgcn_wmma_f32_16x16x32_bf16(false, a1, false, bfr, (short)0, acc1, false, false);
  }

  const float bias  = b2[ncol];
  const int   mbase = (lane >> 4) * 8;
#pragma unroll
  for (int r = 0; r < 8; ++r) {
    const int m = mbase + r;
    Vlog[(size_t)m * V_ + ncol]        = acc0[r] + bias;
    Vlog[(size_t)(m + 16) * V_ + ncol] = acc1[r] + bias;
  }
}

// ---------------------------------------------------------------------------
// K7: out = log_softmax(Vlog, axis=V).  grid B_, 256 threads, 3 passes.
// ---------------------------------------------------------------------------
__global__ void k_vocab_lsm(const float* __restrict__ Vlog, float* __restrict__ o_voc) {
  const int b = blockIdx.x;
  const float* row = Vlog + (size_t)b * V_;
  const int lane = threadIdx.x & 31, wave = threadIdx.x >> 5;
  __shared__ float redm[8], reds[8];
  __shared__ float s_mx, s_lse;

  float mx = -INFINITY;
  for (int v = threadIdx.x; v < V_; v += 256) mx = fmaxf(mx, row[v]);
  mx = wave_max_f(mx);
  if (lane == 0) redm[wave] = mx;
  __syncthreads();
  if (threadIdx.x == 0) {
    float m = -INFINITY;
    for (int i = 0; i < 8; ++i) m = fmaxf(m, redm[i]);
    s_mx = m;
  }
  __syncthreads();
  const float gmx = s_mx;

  float sm = 0.f;
  for (int v = threadIdx.x; v < V_; v += 256) sm += expf(row[v] - gmx);
  sm = wave_sum_f(sm);
  if (lane == 0) reds[wave] = sm;
  __syncthreads();
  if (threadIdx.x == 0) {
    float t = 0.f;
    for (int i = 0; i < 8; ++i) t += reds[i];
    s_lse = gmx + logf(t);
  }
  __syncthreads();
  const float lse = s_lse;

  for (int v = threadIdx.x; v < V_; v += 256) o_voc[(size_t)b * V_ + v] = row[v] - lse;
}

// ---------------------------------------------------------------------------
// launch
// ---------------------------------------------------------------------------
extern "C" void kernel_launch(void* const* d_in, const int* in_sizes, int n_in,
                              void* d_out, int out_size, void* d_ws, size_t ws_size,
                              hipStream_t stream) {
  const float* uh = (const float*)d_in[0];            // (XS,B,D)
  const float* rh = (const float*)d_in[1];            // (YS,B,D)
  const float* kh = (const float*)d_in[2];            // (N,KS,B,D)
  const unsigned char* um = (const unsigned char*)d_in[3];  // (B,XS) bool
  const unsigned char* rm = (const unsigned char*)d_in[4];  // (B,YS) bool
  const unsigned char* km = (const unsigned char*)d_in[5];  // (B,N,KS) bool
  const float* gumbel = (const float*)d_in[6];        // (B,N)
  const float* W1 = (const float*)d_in[7];            // (D,2D)
  const float* b1 = (const float*)d_in[8];            // (D,)
  const float* W2 = (const float*)d_in[9];            // (V,D)
  const float* b2 = (const float*)d_in[10];           // (V,)

  float* ws     = (float*)d_ws;
  float* meanX  = ws;                                  // B*D f32
  float* meanY  = meanX + (size_t)B_ * D_;             // B*D f32
  float* meanK  = meanY + (size_t)B_ * D_;             // B*N*D f32
  float* XYproj = meanK + (size_t)B_ * N_ * D_;        // B*D f32
  float* kidx   = XYproj + (size_t)B_ * D_;            // B*N f32
  __bf16* Svbf  = (__bf16*)(kidx + (size_t)B_ * N_);   // B*D bf16 (64 KB)
  float* Vlog   = (float*)((char*)Svbf + (size_t)B_ * D_ * sizeof(__bf16)); // B*V f32

  float* out     = (float*)d_out;
  float* o_prior = out;                                // B*N
  float* o_post  = out + B_ * N_;                      // B*N
  float* o_kidx  = out + 2 * B_ * N_;                  // B*N
  float* o_sel   = out + 3 * B_ * N_;                  // B*KS*D
  float* o_voc   = o_sel + (size_t)B_ * KS_ * D_;      // B*V

  k_mean_xy   <<<dim3(B_, 2), 256, 0, stream>>>(uh, rh, um, rm, meanX, meanY);
  k_mean_k    <<<dim3(N_, B_), 256, 0, stream>>>(kh, km, meanK);
  k_w1_gemm   <<<dim3(8), 256, 0, stream>>>(meanX, meanY, W1, b1, XYproj);
  k_softmaxes <<<dim3(B_), 256, 0, stream>>>(meanX, XYproj, meanK, gumbel,
                                             o_prior, o_post, o_kidx, kidx, Svbf);
  k_selected  <<<dim3(KS_, B_), 256, 0, stream>>>(kh, km, kidx, o_sel);
  k_w2_gemm   <<<dim3(250), 256, 0, stream>>>(Svbf, W2, b2, Vlog);
  k_vocab_lsm <<<dim3(B_), 256, 0, stream>>>(Vlog, o_voc);
}
